// Model_8237747274039
// MI455X (gfx1250) — matrix operand order repaired and checked by reading, not compiled, not run on hardware
//
#include <hip/hip_runtime.h>
#include <hip/hip_bf16.h>

#define BS   32
#define SEQ  512
#define D    768
#define NE   64
#define NREL 7
#define XST  136   // LDS A-tile row stride (bf16 elems): 128 + 8 pad -> conflict-free
#define YST  772   // LDS y-tile row stride (f32): 768 + 4 pad

typedef __attribute__((ext_vector_type(16))) __bf16 v16bf;
typedef __attribute__((ext_vector_type(8)))  float  v8f;

union Frag16 { v16bf v; uint4 u[2]; };

__device__ __forceinline__ unsigned short f2bf(float f) {
  unsigned int x = __float_as_uint(f);
  x += 0x7FFFu + ((x >> 16) & 1u);          // round-to-nearest-even
  return (unsigned short)(x >> 16);
}

// A fragment 16x32 bf16 from LDS tile (row-major, stride in elems).
// lanes 0-15: row=lane, K 0..7 & 16..23 ; lanes 16-31: row=lane-16, K 8..15 & 24..31
__device__ __forceinline__ v16bf load_a_lds(const unsigned short* tile, int stride) {
  int lane = threadIdx.x & 31;
  int r = lane & 15, hf = lane >> 4;
  const unsigned short* p = tile + r * stride + hf * 8;
  Frag16 f;
  f.u[0] = *(const uint4*)(p);
  f.u[1] = *(const uint4*)(p + 16);
  return f.v;
}

// B fragment 32x16 bf16 where source is stored as [N][K] row-major (weight[n][k]).
// lane = n + 16*khalf; lanes 0-15: K 0..7 & 16..23 ; lanes 16-31: K 8..15 & 24..31 of column n.
__device__ __forceinline__ v16bf load_b_glb(const unsigned short* tile, int stride) {
  int lane = threadIdx.x & 31;
  int n = lane & 15, hf = lane >> 4;
  const unsigned short* p = tile + n * stride + hf * 8;
  Frag16 f;
  f.u[0] = *(const uint4*)(p);
  f.u[1] = *(const uint4*)(p + 16);
  return f.v;
}

// ---------------- Phase 1: span mean-pool -> masked bf16 ent_encode ----------------
__global__ void span_pool_kernel(const float* __restrict__ bert,
                                 const int* __restrict__ spans,
                                 const float* __restrict__ mask,
                                 unsigned short* __restrict__ ent_bf) {
  int e = blockIdx.x, b = blockIdx.y;
  int idx = b * NE + e;
  int u = spans[idx * 2], v = spans[idx * 2 + 1];
  float scale = mask[idx] / (float)(v - u);
  const float* base = bert + (long)b * SEQ * D;
  unsigned short* dst = ent_bf + (long)idx * D;
  for (int s = 0; s < D / 256; ++s) {
    int d = threadIdx.x + 256 * s;
    float acc = 0.f;
    for (int r = u; r < v; ++r) acc += base[(long)r * D + d];
    dst[d] = f2bf(acc * scale);
  }
}

// ---------------- weight fp32 -> bf16 ----------------
__global__ void cvt_bf16_kernel(const float* __restrict__ src,
                                unsigned short* __restrict__ dst, int n) {
  int i = blockIdx.x * blockDim.x + threadIdx.x;
  if (i < n) dst[i] = f2bf(src[i]);
}

// ---------------- Phase 2: q/k projection, 64-row tiles, bf16 WMMA ----------------
__global__ void __launch_bounds__(256) gemm_qk_kernel(
    const unsigned short* __restrict__ ent_bf,
    const unsigned short* __restrict__ w_bf,
    const float* __restrict__ bias,
    float* __restrict__ out) {
  __shared__ __align__(16) unsigned short xs[64 * XST];
  int rb = blockIdx.x * 64;
  int lane = threadIdx.x & 31;
  int ln = lane & 15, hf = lane >> 4;
  int wn = (threadIdx.x >> 5) * 96;  // 8 waves x 96 columns = 768

  const v8f vzero = {};
  v8f acc[4][6];
#pragma unroll
  for (int m = 0; m < 4; ++m)
#pragma unroll
    for (int n = 0; n < 6; ++n) acc[m][n] = vzero;

  int jr = threadIdx.x >> 2, seg = threadIdx.x & 3;
  for (int kc = 0; kc < 6; ++kc) {
    const uint4* src = (const uint4*)(ent_bf + (long)(rb + jr) * D + kc * 128 + seg * 32);
    uint4* dstp = (uint4*)(xs + jr * XST + seg * 32);
#pragma unroll
    for (int g = 0; g < 4; ++g) dstp[g] = src[g];
    __syncthreads();
#pragma unroll
    for (int ks = 0; ks < 4; ++ks) {
      v16bf a[4];
#pragma unroll
      for (int m = 0; m < 4; ++m) a[m] = load_a_lds(xs + m * 16 * XST + ks * 32, XST);
#pragma unroll
      for (int n = 0; n < 6; ++n) {
        v16bf bb = load_b_glb(w_bf + (long)(wn + n * 16) * D + kc * 128 + ks * 32, D);
#pragma unroll
        for (int m = 0; m < 4; ++m)
          acc[m][n] = __builtin_amdgcn_wmma_f32_16x16x32_bf16(
              false, a[m], false, bb, (short)0, acc[m][n], false, false);
      }
    }
    __syncthreads();
  }
#pragma unroll
  for (int m = 0; m < 4; ++m)
#pragma unroll
    for (int n = 0; n < 6; ++n) {
      int col = wn + n * 16 + ln;
      float bv = bias[col];
#pragma unroll
      for (int v = 0; v < 8; ++v) {
        int row = rb + m * 16 + v + 8 * hf;
        out[(long)row * D + col] = acc[m][n][v] + bv;
      }
    }
}

// ---------------- Phase 3: fused pairwise head ----------------
// One block per (b, i): x_j = relu(q_i + k_j) -> GEMM [64x768]x[768x768] (bf16 WMMA)
// -> relu -> [64x768]x[768x7] -> masked out.
union SM {
  unsigned short xs[64 * XST];
  struct { float ys[16 * YST]; float psum[224]; } e;
};

__global__ void __launch_bounds__(256) fused_pair_kernel(
    const float* __restrict__ qbuf,
    const float* __restrict__ kbuf,
    const unsigned short* __restrict__ ln1_bf,
    const float* __restrict__ ln1_b,
    const float* __restrict__ lno_w,
    const float* __restrict__ lno_b,
    const float* __restrict__ mask,
    float* __restrict__ out) {
  __shared__ __align__(16) SM sm;
  int i = blockIdx.x, b = blockIdx.y;
  int lane = threadIdx.x & 31;
  int ln = lane & 15, hf = lane >> 4;
  int wn = (threadIdx.x >> 5) * 96;

  const float* qrow = qbuf + (long)(b * NE + i) * D;
  const float* krows = kbuf + (long)(b * NE) * D;

  const v8f vzero = {};
  v8f acc[4][6];
#pragma unroll
  for (int m = 0; m < 4; ++m)
#pragma unroll
    for (int n = 0; n < 6; ++n) acc[m][n] = vzero;

  int jr = threadIdx.x >> 2, seg = threadIdx.x & 3;
  for (int kc = 0; kc < 6; ++kc) {
    int kb = kc * 128 + seg * 32;
    const float4* qv = (const float4*)(qrow + kb);
    const float4* kv = (const float4*)(krows + (long)jr * D + kb);
    unsigned short* dstp = sm.xs + jr * XST + seg * 32;
#pragma unroll
    for (int g = 0; g < 8; ++g) {
      float4 a = qv[g], c = kv[g];
      float x0 = fmaxf(a.x + c.x, 0.f), x1 = fmaxf(a.y + c.y, 0.f);
      float x2 = fmaxf(a.z + c.z, 0.f), x3 = fmaxf(a.w + c.w, 0.f);
      uint2 pk;
      pk.x = (unsigned)f2bf(x0) | ((unsigned)f2bf(x1) << 16);
      pk.y = (unsigned)f2bf(x2) | ((unsigned)f2bf(x3) << 16);
      *(uint2*)(dstp + g * 4) = pk;
    }
    __syncthreads();
#pragma unroll
    for (int ks = 0; ks < 4; ++ks) {
      v16bf a[4];
#pragma unroll
      for (int m = 0; m < 4; ++m) a[m] = load_a_lds(sm.xs + m * 16 * XST + ks * 32, XST);
#pragma unroll
      for (int n = 0; n < 6; ++n) {
        v16bf bb = load_b_glb(ln1_bf + (long)(wn + n * 16) * D + kc * 128 + ks * 32, D);
#pragma unroll
        for (int m = 0; m < 4; ++m)
          acc[m][n] = __builtin_amdgcn_wmma_f32_16x16x32_bf16(
              false, a[m], false, bb, (short)0, acc[m][n], false, false);
      }
    }
    __syncthreads();
  }

  float mi = mask[b * NE + i];
  for (int m = 0; m < 4; ++m) {
    // stage relu(acc + ln1_b) m-tile [16 x 768] to LDS
#pragma unroll
    for (int n = 0; n < 6; ++n) {
      int col = wn + n * 16 + ln;
      float bv = ln1_b[col];
#pragma unroll
      for (int v = 0; v < 8; ++v) {
        float y = acc[m][n][v] + bv;
        sm.e.ys[(v + 8 * hf) * YST + col] = fmaxf(y, 0.f);
      }
    }
    __syncthreads();
    int t = threadIdx.x;
    if (t < 224) {                     // 16 rows x 7 rels x 2-way split-K
      int pair = t >> 1, h = t & 1;
      int jl = pair / NREL, r = pair - jl * NREL;
      const float* yrow = sm.e.ys + jl * YST + h * 384;
      const float* wrow = lno_w + r * D + h * 384;
      float s = 0.f;
#pragma unroll 4
      for (int c = 0; c < 384; ++c) s += yrow[c] * wrow[c];
      sm.e.psum[t] = s;
    }
    __syncthreads();
    if (t < 112) {
      int jl = t / NREL, r = t - jl * NREL;
      int jj = m * 16 + jl;
      float val = sm.e.psum[2 * t] + sm.e.psum[2 * t + 1] + lno_b[r];
      val *= mi * mask[b * NE + jj];
      out[(long)(((b * NE + i) * NE + jj) * NREL) + r] = val;
    }
    __syncthreads();
  }
}

extern "C" void kernel_launch(void* const* d_in, const int* in_sizes, int n_in,
                              void* d_out, int out_size, void* d_ws, size_t ws_size,
                              hipStream_t stream) {
  const float* bert  = (const float*)d_in[0];
  const int*   spans = (const int*)d_in[1];
  const float* mask  = (const float*)d_in[2];
  const float* wq_w  = (const float*)d_in[3];
  const float* wq_b  = (const float*)d_in[4];
  const float* wk_w  = (const float*)d_in[5];
  const float* wk_b  = (const float*)d_in[6];
  const float* ln1_w = (const float*)d_in[7];
  const float* ln1_b = (const float*)d_in[8];
  const float* lno_w = (const float*)d_in[9];
  const float* lno_b = (const float*)d_in[10];
  float* out = (float*)d_out;

  char* ws = (char*)d_ws;
  unsigned short* ent_bf = (unsigned short*)ws; ws += (size_t)BS * NE * D * 2;
  unsigned short* wq_bf  = (unsigned short*)ws; ws += (size_t)D * D * 2;
  unsigned short* wk_bf  = (unsigned short*)ws; ws += (size_t)D * D * 2;
  unsigned short* ln1_bf = (unsigned short*)ws; ws += (size_t)D * D * 2;
  float* qbuf = (float*)ws;                     ws += (size_t)BS * NE * D * 4;
  float* kbuf = (float*)ws;

  span_pool_kernel<<<dim3(NE, BS), 256, 0, stream>>>(bert, spans, mask, ent_bf);
  int nw = D * D;
  cvt_bf16_kernel<<<(nw + 255) / 256, 256, 0, stream>>>(wq_w, wq_bf, nw);
  cvt_bf16_kernel<<<(nw + 255) / 256, 256, 0, stream>>>(wk_w, wk_bf, nw);
  cvt_bf16_kernel<<<(nw + 255) / 256, 256, 0, stream>>>(ln1_w, ln1_bf, nw);
  gemm_qk_kernel<<<dim3(BS * NE / 64), 256, 0, stream>>>(ent_bf, wq_bf, wq_b, qbuf);
  gemm_qk_kernel<<<dim3(BS * NE / 64), 256, 0, stream>>>(ent_bf, wk_bf, wk_b, kbuf);
  fused_pair_kernel<<<dim3(NE, BS), 256, 0, stream>>>(qbuf, kbuf, ln1_bf, ln1_b,
                                                      lno_w, lno_b, mask, out);
}
